// ImprovedWav2Vec2CQCCDetector_4423816315063
// MI455X (gfx1250) — compile-verified, matrix operand order checked by reading, not run on hardware
//
#include <hip/hip_runtime.h>
#include <math.h>

// ---------------- types for CDNA5 WMMA ----------------
typedef __bf16 bf16;
typedef __bf16 v16bf __attribute__((ext_vector_type(16)));
typedef float  v8f   __attribute__((ext_vector_type(8)));

union FragB16 { v16bf v; uint4 u[2]; };

// Model dims
#define Dm 768
#define Hh 4
#define HD 192
#define Bb 8
#define Nn 1024
#define Tt 2000

__device__ __forceinline__ float gelu_f(float x) {
    return 0.5f * x * (1.0f + erff(x * 0.70710678118654752f));
}

// =====================================================================
// WMMA GEMM:  C[M,N] = A[M,K](bf16, row-major) * W[N,K](bf16, row-major)^T + bias[N]
// block = 128 threads (4 waves); wave tile = 32(M) x 64(N); block tile 128 x 64
// grid = (N/64, M/128, batch)
// Requires: M % 128 == 0, N % 64 == 0, K % 32 == 0 (all call sites satisfy this)
// =====================================================================
__global__ __launch_bounds__(128)
void gemm_bf16_nt(const bf16* __restrict__ A, const bf16* __restrict__ W,
                  const float* __restrict__ bias, float* __restrict__ C,
                  int M, int N, int K,
                  long long sA, long long sW, long long sC)
{
    const int lane = threadIdx.x & 31;
    const int wave = threadIdx.x >> 5;
    A += (long long)blockIdx.z * sA;
    W += (long long)blockIdx.z * sW;
    C += (long long)blockIdx.z * sC;

    const int row0 = (blockIdx.y * 4 + wave) * 32;  // 32 rows per wave
    const int col0 = blockIdx.x * 64;               // 64 cols per wave

    const int mlo   = lane & 15;
    const int khalf = lane >> 4;                    // 0/1 -> K offset 0/8 (per ISA 16-bit layout)

    const bf16* a0 = A + (long long)(row0 + mlo) * K + khalf * 8;
    const bf16* a1 = a0 + 16LL * K;
    const bf16* w0 = W + (long long)(col0 + mlo) * K + khalf * 8;

    v8f acc[2][4] = {};

    for (int k = 0; k < K; k += 32) {
        FragB16 fa[2], fb[4];
        {
            const bf16* p = a0 + k;
            fa[0].u[0] = *(const uint4*)(p);
            fa[0].u[1] = *(const uint4*)(p + 16);
            p = a1 + k;
            fa[1].u[0] = *(const uint4*)(p);
            fa[1].u[1] = *(const uint4*)(p + 16);
        }
#pragma unroll
        for (int j = 0; j < 4; ++j) {
            const bf16* p = w0 + (long long)(16 * j) * K + k;
            fb[j].u[0] = *(const uint4*)(p);
            fb[j].u[1] = *(const uint4*)(p + 16);
        }
#pragma unroll
        for (int i = 0; i < 2; ++i)
#pragma unroll
            for (int j = 0; j < 4; ++j)
                acc[i][j] = __builtin_amdgcn_wmma_f32_16x16x32_bf16(
                    false, fa[i].v, false, fb[j].v, (short)0, acc[i][j], false, false);
    }

    // D layout: lanes 0-15 -> rows m0..m0+7 (VGPR r = row r), lanes 16-31 -> rows m0+8..m0+15
    const int crow = (lane >> 4) * 8;
    const int ccol = lane & 15;
#pragma unroll
    for (int i = 0; i < 2; ++i) {
#pragma unroll
        for (int j = 0; j < 4; ++j) {
            const int cc = col0 + j * 16 + ccol;
            const float bv = bias ? bias[cc] : 0.0f;
            long long base = (long long)(row0 + i * 16 + crow) * N + cc;
#pragma unroll
            for (int r = 0; r < 8; ++r)
                C[base + (long long)r * N] = acc[i][j][r] + bv;
        }
    }
}

// =====================================================================
// conv1 (20ch -> 128ch, k=3, pad=1) + BN + GELU, direct
// =====================================================================
__global__ void k_conv1(const float* __restrict__ x, const float* __restrict__ w,
                        const float* __restrict__ cb,
                        const float* __restrict__ g, const float* __restrict__ be,
                        const float* __restrict__ mu, const float* __restrict__ var,
                        float* __restrict__ y)
{
    int idx = blockIdx.x * blockDim.x + threadIdx.x;
    if (idx >= Bb * 128 * Tt) return;
    int t = idx % Tt;
    int co = (idx / Tt) % 128;
    int b = idx / (Tt * 128);
    float s = cb[co];
    for (int ci = 0; ci < 20; ++ci) {
        const float* xr = x + ((long long)b * 20 + ci) * Tt;
        const float* wr = w + ((long long)co * 20 + ci) * 3;
#pragma unroll
        for (int dx = 0; dx < 3; ++dx) {
            int tt = t + dx - 1;
            if (tt >= 0 && tt < Tt) s += wr[dx] * xr[tt];
        }
    }
    s = (s - mu[co]) * rsqrtf(var[co] + 1e-5f) * g[co] + be[co];
    y[idx] = gelu_f(s);
}

// im2col for conv2: A2[b*T + t, cin*3+dx] (bf16) from x1[b, cin, t+dx-1]
__global__ void k_im2col(const float* __restrict__ x1, bf16* __restrict__ a2)
{
    long long idx = (long long)blockIdx.x * blockDim.x + threadIdx.x;
    if (idx >= (long long)Bb * Tt * 384) return;
    int col = (int)(idx % 384);
    long long row = idx / 384;
    int t = (int)(row % Tt);
    int b = (int)(row / Tt);
    int cin = col / 3, dx = col % 3;
    int tt = t + dx - 1;
    float v = 0.0f;
    if (tt >= 0 && tt < Tt) v = x1[((long long)b * 128 + cin) * Tt + tt];
    a2[idx] = (bf16)v;
}

// per-column BN + GELU over [rows, 768]
__global__ void k_bn_gelu(float* __restrict__ x,
                          const float* __restrict__ g, const float* __restrict__ be,
                          const float* __restrict__ mu, const float* __restrict__ var,
                          long long n)
{
    long long idx = (long long)blockIdx.x * blockDim.x + threadIdx.x;
    if (idx >= n) return;
    int c = (int)(idx % Dm);
    float s = (x[idx] - mu[c]) * rsqrtf(var[c] + 1e-5f) * g[c] + be[c];
    x[idx] = gelu_f(s);
}

// cq[b,n,d] = lerp(x2 rows) + pos_embed[n,d]   (x2 is [B*T, 768])
__global__ void k_build_cq(const float* __restrict__ x2, const float* __restrict__ pos,
                           float* __restrict__ cq)
{
    long long idx = (long long)blockIdx.x * blockDim.x + threadIdx.x;
    if (idx >= (long long)Bb * Nn * Dm) return;
    int d = (int)(idx % Dm);
    int n = (int)((idx / Dm) % Nn);
    int b = (int)(idx / ((long long)Dm * Nn));
    float p = (n + 0.5f) * ((float)Tt / (float)Nn) - 0.5f;
    p = fminf(fmaxf(p, 0.0f), (float)(Tt - 1));
    int lo = (int)floorf(p);
    int hi = min(lo + 1, Tt - 1);
    float wgt = p - (float)lo;
    float v = x2[((long long)b * Tt + lo) * Dm + d] * (1.0f - wgt)
            + x2[((long long)b * Tt + hi) * Dm + d] * wgt;
    cq[idx] = v + pos[(long long)n * Dm + d];
}

__global__ void k_build_w(const float* __restrict__ w2v, const float* __restrict__ pos,
                          float* __restrict__ o)
{
    long long idx = (long long)blockIdx.x * blockDim.x + threadIdx.x;
    if (idx >= (long long)Bb * Nn * Dm) return;
    int d = (int)(idx % Dm);
    int n = (int)((idx / Dm) % Nn);
    o[idx] = w2v[idx] + pos[(long long)n * Dm + d];
}

// warp-per-row layernorm (row len 768) -> bf16
__global__ void k_ln_bf16(const float* __restrict__ X, const float* __restrict__ g,
                          const float* __restrict__ be, bf16* __restrict__ O, int rows)
{
    int warp = (blockIdx.x * blockDim.x + threadIdx.x) >> 5;
    int lane = threadIdx.x & 31;
    if (warp >= rows) return;
    const float* x = X + (long long)warp * Dm;
    float s = 0.f, s2 = 0.f;
#pragma unroll
    for (int i = 0; i < 24; ++i) { float v = x[lane + i * 32]; s += v; s2 += v * v; }
#pragma unroll
    for (int off = 16; off; off >>= 1) { s += __shfl_xor(s, off, 32); s2 += __shfl_xor(s2, off, 32); }
    float mean = s * (1.0f / Dm);
    float inv = rsqrtf(s2 * (1.0f / Dm) - mean * mean + 1e-5f);
    bf16* o = O + (long long)warp * Dm;
#pragma unroll
    for (int i = 0; i < 24; ++i) {
        int c = lane + i * 32;
        o[c] = (bf16)((x[c] - mean) * inv * g[c] + be[c]);
    }
}

// split Q [8192,768] -> qh bf16 [B,H,N,HD]
__global__ void k_split_q(const float* __restrict__ Q, bf16* __restrict__ qh)
{
    long long idx = (long long)blockIdx.x * blockDim.x + threadIdx.x;
    if (idx >= (long long)Bb * Nn * Dm) return;
    int col = (int)(idx % Dm);
    long long row = idx / Dm;
    int n = (int)(row & 1023), b = (int)(row >> 10);
    int h = col / HD, d = col % HD;
    qh[(((long long)(b * Hh + h) << 10) + n) * HD + d] = (bf16)Q[idx];
}

// split KV [8192,1536] -> kh bf16 [B,H,N,HD], vhT bf16 [B,H,HD,N]
__global__ void k_split_kv(const float* __restrict__ KV, bf16* __restrict__ kh,
                           bf16* __restrict__ vhT)
{
    long long idx = (long long)blockIdx.x * blockDim.x + threadIdx.x;
    if (idx >= (long long)Bb * Nn * Dm) return;
    int col = (int)(idx % Dm);
    long long row = idx / Dm;
    int n = (int)(row & 1023), b = (int)(row >> 10);
    int h = col / HD, d = col % HD;
    kh[(((long long)(b * Hh + h) << 10) + n) * HD + d] = (bf16)KV[row * 1536 + col];
    vhT[((long long)(b * Hh + h) * HD + d) * Nn + n] = (bf16)KV[row * 1536 + Dm + col];
}

// warp-per-row softmax over 1024 cols, f32 in -> bf16 out (pre-scale)
__global__ void k_softmax_bf16(const float* __restrict__ S, bf16* __restrict__ A,
                               int rows, float scale)
{
    int warp = (blockIdx.x * blockDim.x + threadIdx.x) >> 5;
    int lane = threadIdx.x & 31;
    if (warp >= rows) return;
    const float* s = S + (long long)warp * 1024;
    float v[32];
    float mx = -1e30f;
#pragma unroll
    for (int i = 0; i < 32; ++i) { v[i] = s[lane + i * 32] * scale; mx = fmaxf(mx, v[i]); }
#pragma unroll
    for (int off = 16; off; off >>= 1) mx = fmaxf(mx, __shfl_xor(mx, off, 32));
    float sum = 0.f;
#pragma unroll
    for (int i = 0; i < 32; ++i) { v[i] = __expf(v[i] - mx); sum += v[i]; }
#pragma unroll
    for (int off = 16; off; off >>= 1) sum += __shfl_xor(sum, off, 32);
    float inv = 1.0f / sum;
    bf16* a = A + (long long)warp * 1024;
#pragma unroll
    for (int i = 0; i < 32; ++i) a[lane + i * 32] = (bf16)(v[i] * inv);
}

// warp-per-row softmax over 1024 cols, f32 -> f32
__global__ void k_softmax_f32(const float* __restrict__ S, float* __restrict__ P, int rows)
{
    int warp = (blockIdx.x * blockDim.x + threadIdx.x) >> 5;
    int lane = threadIdx.x & 31;
    if (warp >= rows) return;
    const float* s = S + (long long)warp * 1024;
    float v[32];
    float mx = -1e30f;
#pragma unroll
    for (int i = 0; i < 32; ++i) { v[i] = s[lane + i * 32]; mx = fmaxf(mx, v[i]); }
#pragma unroll
    for (int off = 16; off; off >>= 1) mx = fmaxf(mx, __shfl_xor(mx, off, 32));
    float sum = 0.f;
#pragma unroll
    for (int i = 0; i < 32; ++i) { v[i] = __expf(v[i] - mx); sum += v[i]; }
#pragma unroll
    for (int off = 16; off; off >>= 1) sum += __shfl_xor(sum, off, 32);
    float inv = 1.0f / sum;
    float* p = P + (long long)warp * 1024;
#pragma unroll
    for (int i = 0; i < 32; ++i) p[lane + i * 32] = v[i] * inv;
}

// merge heads: oh f32 [B,H,N,HD] -> obf bf16 [B*N, D]
__global__ void k_merge(const float* __restrict__ oh, bf16* __restrict__ obf)
{
    long long idx = (long long)blockIdx.x * blockDim.x + threadIdx.x;
    if (idx >= (long long)Bb * Nn * Dm) return;
    int col = (int)(idx % Dm);
    long long row = idx / Dm;
    int n = (int)(row & 1023), b = (int)(row >> 10);
    int h = col / HD, d = col % HD;
    obf[idx] = (bf16)oh[(((long long)(b * Hh + h) << 10) + n) * HD + d];
}

__global__ void k_add(const float* __restrict__ a, const float* __restrict__ b,
                      float* __restrict__ o, long long n)
{
    long long idx = (long long)blockIdx.x * blockDim.x + threadIdx.x;
    if (idx < n) o[idx] = a[idx] + b[idx];
}

__global__ void k_f32_to_bf16(const float* __restrict__ s, bf16* __restrict__ d, long long n)
{
    long long idx = (long long)blockIdx.x * blockDim.x + threadIdx.x;
    if (idx < n) d[idx] = (bf16)s[idx];
}

// s_j[row] = dot(h[row,:], aw[:])   (warp per row, 768)
__global__ void k_gat_sj(const float* __restrict__ H, const float* __restrict__ aw,
                         float* __restrict__ sj, int rows)
{
    int warp = (blockIdx.x * blockDim.x + threadIdx.x) >> 5;
    int lane = threadIdx.x & 31;
    if (warp >= rows) return;
    const float* h = H + (long long)warp * Dm;
    float s = 0.f;
#pragma unroll
    for (int i = 0; i < 24; ++i) { int c = lane + i * 32; s += h[c] * aw[c]; }
#pragma unroll
    for (int off = 16; off; off >>= 1) s += __shfl_xor(s, off, 32);
    if (lane == 0) sj[warp] = s;
}

// ctx[b,d] = sum_n p[b,n] * h[b,n,d]   grid(3, B), block 256
__global__ void k_gat_ctx(const float* __restrict__ P, const float* __restrict__ H,
                          float* __restrict__ ctx)
{
    int d = blockIdx.x * blockDim.x + threadIdx.x;
    int b = blockIdx.y;
    if (d >= Dm) return;
    float s = 0.f;
    for (int n = 0; n < Nn; ++n)
        s += P[b * Nn + n] * H[(((long long)b << 10) + n) * Dm + d];
    ctx[b * Dm + d] = s;
}

// x[row,:] = LN(ctx[b,:] + x[row,:]) in place; also write bf16 copy
__global__ void k_res_ln(float* __restrict__ X, const float* __restrict__ ctx,
                         const float* __restrict__ g, const float* __restrict__ be,
                         bf16* __restrict__ O, int rows)
{
    int warp = (blockIdx.x * blockDim.x + threadIdx.x) >> 5;
    int lane = threadIdx.x & 31;
    if (warp >= rows) return;
    int b = warp >> 10;
    float* x = X + (long long)warp * Dm;
    const float* cx = ctx + b * Dm;
    float v[24];
    float s = 0.f, s2 = 0.f;
#pragma unroll
    for (int i = 0; i < 24; ++i) {
        int c = lane + i * 32;
        v[i] = x[c] + cx[c];
        s += v[i]; s2 += v[i] * v[i];
    }
#pragma unroll
    for (int off = 16; off; off >>= 1) { s += __shfl_xor(s, off, 32); s2 += __shfl_xor(s2, off, 32); }
    float mean = s * (1.0f / Dm);
    float inv = rsqrtf(s2 * (1.0f / Dm) - mean * mean + 1e-5f);
    bf16* o = O + (long long)warp * Dm;
#pragma unroll
    for (int i = 0; i < 24; ++i) {
        int c = lane + i * 32;
        float r = (v[i] - mean) * inv * g[c] + be[c];
        x[c] = r;
        o[c] = (bf16)r;
    }
}

// pooled[b,d] = mean_n x[b,n,d]    grid(3, B), block 256
__global__ void k_meanpool(const float* __restrict__ X, float* __restrict__ P)
{
    int d = blockIdx.x * blockDim.x + threadIdx.x;
    int b = blockIdx.y;
    if (d >= Dm) return;
    float s = 0.f;
    for (int n = 0; n < Nn; ++n) s += X[(((long long)b << 10) + n) * Dm + d];
    P[b * Dm + d] = s * (1.0f / Nn);
}

// classifier: block per b, 128 threads
__global__ void k_classifier(const float* __restrict__ pooled,
                             const float* __restrict__ w1, const float* __restrict__ b1,
                             const float* __restrict__ w2, const float* __restrict__ b2,
                             float* __restrict__ out)
{
    __shared__ float h1[128];
    int b = blockIdx.x, j = threadIdx.x;
    const float* p = pooled + b * Dm;
    float s = b1[j];
    const float* wr = w1 + (long long)j * Dm;
    for (int k = 0; k < Dm; ++k) s += p[k] * wr[k];
    h1[j] = gelu_f(s);
    __syncthreads();
    if (j < 2) {
        float o = b2[j];
        const float* w2r = w2 + j * 128;
        for (int k = 0; k < 128; ++k) o += h1[k] * w2r[k];
        out[b * 2 + j] = o;
    }
}

// =====================================================================
extern "C" void kernel_launch(void* const* d_in, const int* in_sizes, int n_in,
                              void* d_out, int out_size, void* d_ws, size_t ws_size,
                              hipStream_t stream)
{
    const float* w2v     = (const float*)d_in[0];
    const float* cqcc    = (const float*)d_in[1];
    const float* conv1_w = (const float*)d_in[2];
    const float* conv1_b = (const float*)d_in[3];
    const float* bn1_g = (const float*)d_in[4], *bn1_b = (const float*)d_in[5];
    const float* bn1_m = (const float*)d_in[6], *bn1_v = (const float*)d_in[7];
    const float* conv2_w = (const float*)d_in[8];
    const float* conv2_b = (const float*)d_in[9];
    const float* bn2_g = (const float*)d_in[10], *bn2_b = (const float*)d_in[11];
    const float* bn2_m = (const float*)d_in[12], *bn2_v = (const float*)d_in[13];
    const float* pos   = (const float*)d_in[14];
    const float* lnq_g = (const float*)d_in[15], *lnq_b = (const float*)d_in[16];
    const float* lnkv_g = (const float*)d_in[17], *lnkv_b = (const float*)d_in[18];
    const float* a_in_w[2]  = { (const float*)d_in[19], (const float*)d_in[23] };
    const float* a_in_b[2]  = { (const float*)d_in[20], (const float*)d_in[24] };
    const float* a_out_w[2] = { (const float*)d_in[21], (const float*)d_in[25] };
    const float* a_out_b[2] = { (const float*)d_in[22], (const float*)d_in[26] };
    const float* gat_fc_w = (const float*)d_in[27];
    const float* gat_fc_b = (const float*)d_in[28];
    const float* gat_a_w  = (const float*)d_in[29];
    const float* gln_g = (const float*)d_in[31], *gln_b = (const float*)d_in[32];
    const float* clf_w1 = (const float*)d_in[33], *clf_b1 = (const float*)d_in[34];
    const float* clf_w2 = (const float*)d_in[35], *clf_b2 = (const float*)d_in[36];
    float* out = (float*)d_out;

    // ---------- workspace layout ----------
    char* base = (char*)d_ws;
    size_t off = 0;
    auto alloc = [&](size_t bytes) -> char* {
        char* p = base + off;
        off += (bytes + 255) & ~(size_t)255;
        return p;
    };
    const long long BND = (long long)Bb * Nn * Dm;   // 6,291,456

    float* x1    = (float*)alloc((size_t)Bb * 128 * Tt * 4);
    bf16*  A2    = (bf16*) alloc((size_t)Bb * Tt * 384 * 2);
    float* x2    = (float*)alloc((size_t)Bb * Tt * Dm * 4);
    float* cq    = (float*)alloc(BND * 4);
    float* wv    = (float*)alloc(BND * 4);
    bf16*  lnq   = (bf16*) alloc(BND * 2);
    bf16*  lnkv  = (bf16*) alloc(BND * 2);
    float* Qbuf  = (float*)alloc(BND * 4);
    float* KVbuf = (float*)alloc(BND * 2 * 4);
    bf16*  qh    = (bf16*) alloc(BND * 2);
    bf16*  kh    = (bf16*) alloc(BND * 2);
    bf16*  vhT   = (bf16*) alloc(BND * 2);
    float* scores= (float*)alloc((size_t)Hh * Nn * Nn * 4);   // per-b
    bf16*  alpha = (bf16*) alloc((size_t)Hh * Nn * Nn * 2);
    float* oh    = (float*)alloc(BND * 4);
    bf16*  obf   = (bf16*) alloc(BND * 2);
    float* out1  = (float*)alloc(BND * 4);
    float* out2  = (float*)alloc(BND * 4);
    float* xsum  = (float*)alloc(BND * 4);
    bf16*  xbf   = (bf16*) alloc(BND * 2);
    float* hbuf  = (float*)alloc(BND * 4);
    float* sj    = (float*)alloc((size_t)Bb * Nn * 4);
    float* pbuf  = (float*)alloc((size_t)Bb * Nn * 4);
    float* ctx   = (float*)alloc((size_t)Bb * Dm * 4);
    float* pooled= (float*)alloc((size_t)Bb * Dm * 4);
    bf16*  w2bf  = (bf16*) alloc((size_t)Dm * 384 * 2);
    bf16*  winbf = (bf16*) alloc((size_t)3 * Dm * Dm * 2);
    bf16*  woutbf= (bf16*) alloc((size_t)Dm * Dm * 2);
    bf16*  fcbf  = (bf16*) alloc((size_t)Dm * Dm * 2);
    (void)ws_size; (void)n_in; (void)in_sizes; (void)out_size;

    auto blocks = [](long long n, int bs) { return (unsigned)((n + bs - 1) / bs); };

    // ---------- conv front-end ----------
    k_f32_to_bf16<<<blocks((long long)Dm * 384, 256), 256, 0, stream>>>(conv2_w, w2bf, (long long)Dm * 384);
    k_conv1<<<blocks((long long)Bb * 128 * Tt, 256), 256, 0, stream>>>(
        cqcc, conv1_w, conv1_b, bn1_g, bn1_b, bn1_m, bn1_v, x1);
    k_im2col<<<blocks((long long)Bb * Tt * 384, 256), 256, 0, stream>>>(x1, A2);
    gemm_bf16_nt<<<dim3(Dm / 64, (Bb * Tt) / 128, 1), 128, 0, stream>>>(
        A2, w2bf, conv2_b, x2, Bb * Tt, Dm, 384, 0, 0, 0);
    k_bn_gelu<<<blocks((long long)Bb * Tt * Dm, 256), 256, 0, stream>>>(
        x2, bn2_g, bn2_b, bn2_m, bn2_v, (long long)Bb * Tt * Dm);
    k_build_cq<<<blocks(BND, 256), 256, 0, stream>>>(x2, pos, cq);
    k_build_w<<<blocks(BND, 256), 256, 0, stream>>>(w2v, pos, wv);

    // ---------- two cross-attention blocks ----------
    const float scale = 0.07216878364870323f;  // 1/sqrt(192)
    for (int m = 0; m < 2; ++m) {
        const float* qsrc  = (m == 0) ? cq : wv;
        const float* kvsrc = (m == 0) ? wv : cq;
        float* outm = (m == 0) ? out1 : out2;

        k_f32_to_bf16<<<blocks((long long)3 * Dm * Dm, 256), 256, 0, stream>>>(
            a_in_w[m], winbf, (long long)3 * Dm * Dm);
        k_f32_to_bf16<<<blocks((long long)Dm * Dm, 256), 256, 0, stream>>>(
            a_out_w[m], woutbf, (long long)Dm * Dm);

        k_ln_bf16<<<blocks((long long)Bb * Nn * 32, 128), 128, 0, stream>>>(qsrc, lnq_g, lnq_b, lnq, Bb * Nn);
        k_ln_bf16<<<blocks((long long)Bb * Nn * 32, 128), 128, 0, stream>>>(kvsrc, lnkv_g, lnkv_b, lnkv, Bb * Nn);

        // Q = lnq @ Wq^T ; KV = lnkv @ Wkv^T   (row slices of in_w)
        gemm_bf16_nt<<<dim3(Dm / 64, (Bb * Nn) / 128, 1), 128, 0, stream>>>(
            lnq, winbf, a_in_b[m], Qbuf, Bb * Nn, Dm, Dm, 0, 0, 0);
        gemm_bf16_nt<<<dim3((2 * Dm) / 64, (Bb * Nn) / 128, 1), 128, 0, stream>>>(
            lnkv, winbf + (size_t)Dm * Dm, a_in_b[m] + Dm, KVbuf, Bb * Nn, 2 * Dm, Dm, 0, 0, 0);

        k_split_q<<<blocks(BND, 256), 256, 0, stream>>>(Qbuf, qh);
        k_split_kv<<<blocks(BND, 256), 256, 0, stream>>>(KVbuf, kh, vhT);

        for (int b = 0; b < Bb; ++b) {
            const bf16* qh_b  = qh  + (size_t)b * Hh * Nn * HD;
            const bf16* kh_b  = kh  + (size_t)b * Hh * Nn * HD;
            const bf16* vhT_b = vhT + (size_t)b * Hh * HD * Nn;
            float* oh_b = oh + (size_t)b * Hh * Nn * HD;
            // scores[h] = qh[h] @ kh[h]^T     (batched over H)
            gemm_bf16_nt<<<dim3(Nn / 64, Nn / 128, Hh), 128, 0, stream>>>(
                qh_b, kh_b, nullptr, scores, Nn, Nn, HD,
                (long long)Nn * HD, (long long)Nn * HD, (long long)Nn * Nn);
            k_softmax_bf16<<<blocks((long long)Hh * Nn * 32, 128), 128, 0, stream>>>(
                scores, alpha, Hh * Nn, scale);
            // oh[h] = alpha[h] @ V[h]   (V^T stored as [HD, N])
            gemm_bf16_nt<<<dim3(HD / 64, Nn / 128, Hh), 128, 0, stream>>>(
                alpha, vhT_b, nullptr, oh_b, Nn, HD, Nn,
                (long long)Nn * Nn, (long long)HD * Nn, (long long)Nn * HD);
        }
        k_merge<<<blocks(BND, 256), 256, 0, stream>>>(oh, obf);
        gemm_bf16_nt<<<dim3(Dm / 64, (Bb * Nn) / 128, 1), 128, 0, stream>>>(
            obf, woutbf, a_out_b[m], outm, Bb * Nn, Dm, Dm, 0, 0, 0);
    }

    // ---------- GAT stack (softmax over j of s_i+s_j+b == softmax(s_j): rank-1 context) ----------
    k_add<<<blocks(BND, 256), 256, 0, stream>>>(out1, out2, xsum, BND);
    k_f32_to_bf16<<<blocks(BND, 256), 256, 0, stream>>>(xsum, xbf, BND);

    for (int l = 0; l < 3; ++l) {
        k_f32_to_bf16<<<blocks((long long)Dm * Dm, 256), 256, 0, stream>>>(
            gat_fc_w + (size_t)l * Dm * Dm, fcbf, (long long)Dm * Dm);
        gemm_bf16_nt<<<dim3(Dm / 64, (Bb * Nn) / 128, 1), 128, 0, stream>>>(
            xbf, fcbf, gat_fc_b + (size_t)l * Dm, hbuf, Bb * Nn, Dm, Dm, 0, 0, 0);
        k_gat_sj<<<blocks((long long)Bb * Nn * 32, 128), 128, 0, stream>>>(
            hbuf, gat_a_w + (size_t)l * 2 * Dm + Dm, sj, Bb * Nn);
        k_softmax_f32<<<blocks((long long)Bb * 32, 128), 128, 0, stream>>>(sj, pbuf, Bb);
        k_gat_ctx<<<dim3(3, Bb), 256, 0, stream>>>(pbuf, hbuf, ctx);
        k_res_ln<<<blocks((long long)Bb * Nn * 32, 128), 128, 0, stream>>>(
            xsum, ctx, gln_g + (size_t)l * Dm, gln_b + (size_t)l * Dm, xbf, Bb * Nn);
    }

    // ---------- pool + classifier ----------
    k_meanpool<<<dim3(3, Bb), 256, 0, stream>>>(xsum, pooled);
    k_classifier<<<Bb, 128, 0, stream>>>(pooled, clf_w1, clf_b1, clf_w2, clf_b2, out);
}
